// MultiHeadedAttention_33492154974322
// MI455X (gfx1250) — matrix operand order repaired and checked by reading, not compiled, not run on hardware
//
#include <hip/hip_runtime.h>

typedef __bf16 bf16;
typedef __attribute__((ext_vector_type(16))) __bf16 v16bf;
typedef __attribute__((ext_vector_type(8)))  __bf16 v8bf;
typedef __attribute__((ext_vector_type(8)))  float  v8f;

#define BB 16
#define TT 512
#define SZ 512
#define HH 8
#define HD 64

__device__ __forceinline__ bf16 f2bf(float f) {
  unsigned u = __builtin_bit_cast(unsigned, f);
  u += 0x7FFFu + ((u >> 16) & 1u);               // round-to-nearest-even
  unsigned short h = (unsigned short)(u >> 16);
  return __builtin_bit_cast(bf16, h);
}
__device__ __forceinline__ float bf2f(bf16 b) {
  unsigned short h = __builtin_bit_cast(unsigned short, b);
  unsigned u = ((unsigned)h) << 16;
  return __builtin_bit_cast(float, u);
}
__device__ __forceinline__ v16bf cat8(v8bf a, v8bf b) {
  return __builtin_shufflevector(a, b, 0,1,2,3,4,5,6,7,8,9,10,11,12,13,14,15);
}
__device__ __forceinline__ v8f wmma_bf16(v16bf a, v16bf b, v8f c) {
  return __builtin_amdgcn_wmma_f32_16x16x32_bf16(false, a, false, b, (short)0, c, false, false);
}

// LDS aperture places the LDS byte offset in addr[31:0] (ISA 10.2 aperture map),
// so truncating a generic pointer to a __shared__ object yields the DS address.
__device__ __forceinline__ unsigned ldsOff(const void* p) {
  return (unsigned)(unsigned long long)p;
}
// CDNA5 async global->LDS copy (ASYNCcnt-tracked). vdst = LDS byte address,
// vaddr = 64-bit global address.
__device__ __forceinline__ void async_b128(unsigned lds, const void* g) {
  unsigned long long ga = (unsigned long long)g;
  asm volatile("global_load_async_to_lds_b128 %0, %1, off"
               :: "v"(lds), "v"(ga) : "memory");
}
#define WAIT_ASYNC(n) asm volatile("s_wait_asynccnt " #n ::: "memory")

// ---------------------------------------------------------------------------
// f32 -> bf16 conversion (one-shot; halves GEMM read traffic)
// ---------------------------------------------------------------------------
__global__ __launch_bounds__(256) void cvt_bf16_kernel(
    const float* __restrict__ src, bf16* __restrict__ dst, int n)
{
  const int i = (blockIdx.x * 256 + threadIdx.x) * 8;
  if (i >= n) return;
  float4 a = *(const float4*)(src + i);
  float4 b = *(const float4*)(src + i + 4);
  v8bf o;
  o[0]=f2bf(a.x); o[1]=f2bf(a.y); o[2]=f2bf(a.z); o[3]=f2bf(a.w);
  o[4]=f2bf(b.x); o[5]=f2bf(b.y); o[6]=f2bf(b.z); o[7]=f2bf(b.w);
  *(v8bf*)(dst + i) = o;
}

// ---------------------------------------------------------------------------
// GEMM: C[M,N] = X[M,K] @ W[N,K]^T + bias[N]   (bf16 in via async-to-LDS,
// f32 out). Block tile 128x64, 8 waves each owning 32x32 (4 WMMA/K-step),
// K-step 32, double-buffered LDS fed by global_load_async_to_lds_b128.
// ---------------------------------------------------------------------------
__global__ __launch_bounds__(256) void gemm_bf16_kernel(
    const bf16* __restrict__ X, const bf16* __restrict__ W,
    const float* __restrict__ bias, float* __restrict__ C,
    int M, int N, int K)
{
  __shared__ __align__(16) bf16 As[2][128 * 40];   // [buf][row m][k] stride 40
  __shared__ __align__(16) bf16 Bs[2][64 * 40];    // [buf][col n][k] stride 40
  const int m0 = blockIdx.y * 128, n0 = blockIdx.x * 64;
  const int tid = threadIdx.x, lane = tid & 31, w = tid >> 5;
  const int hl = lane >> 4, l16 = lane & 15;
  const int mt = w & 3, nh = w >> 2;

  // Async staging: A has 512 16-byte segments (2/thread), B has 256 (1/thread)
  const int rA0 = tid >> 2,         cA0 = (tid & 3) * 8;
  const int rA1 = (tid + 256) >> 2, cA1 = cA0;
  const int rB  = tid >> 2,         cB  = cA0;
  const unsigned baseA = ldsOff(&As[0][0]);
  const unsigned baseB = ldsOff(&Bs[0][0]);
  const unsigned ofsA0 = (unsigned)((rA0 * 40 + cA0) * 2);
  const unsigned ofsA1 = (unsigned)((rA1 * 40 + cA1) * 2);
  const unsigned ofsB  = (unsigned)((rB  * 40 + cB ) * 2);
  const bf16* gA0 = X + (size_t)(m0 + rA0) * K + cA0;
  const bf16* gA1 = X + (size_t)(m0 + rA1) * K + cA1;
  const bf16* gB  = W + (size_t)(n0 + rB ) * K + cB;

  auto issueStage = [&](int k0, int buf) {
    const unsigned ba = baseA + (unsigned)buf * (128u * 40u * 2u);
    const unsigned bb = baseB + (unsigned)buf * (64u * 40u * 2u);
    async_b128(ba + ofsA0, gA0 + k0);
    async_b128(ba + ofsA1, gA1 + k0);
    async_b128(bb + ofsB,  gB  + k0);
    __builtin_prefetch(gA0 + k0 + 64, 0, 3);     // global_prefetch two stages out
  };

  v8f acc00 = {}, acc01 = {}, acc10 = {}, acc11 = {};
  const int NK = K / 32;
  issueStage(0, 0);
  for (int ks = 0; ks < NK; ks++) {
    const int buf = ks & 1;
    __syncthreads();                       // all reads of buf^1 complete
    if (ks + 1 < NK) { issueStage((ks + 1) * 32, buf ^ 1); WAIT_ASYNC(3); }
    else             { WAIT_ASYNC(0); }
    __syncthreads();                       // stage ks resident in LDS for all waves

    const bf16* as = As[buf];
    const bf16* bs = Bs[buf];
    const bf16* ap0 = as + (mt * 32 + l16) * 40;
    const bf16* ap1 = as + (mt * 32 + 16 + l16) * 40;
    v16bf A0 = cat8(*(const v8bf*)(ap0 + 16 * hl), *(const v8bf*)(ap0 + 16 * hl + 8));
    v16bf A1 = cat8(*(const v8bf*)(ap1 + 16 * hl), *(const v8bf*)(ap1 + 16 * hl + 8));
    const bf16* bp0 = bs + (nh * 32 + l16) * 40 + 16 * hl;
    const bf16* bp1 = bs + (nh * 32 + 16 + l16) * 40 + 16 * hl;
    v16bf B0 = cat8(*(const v8bf*)bp0, *(const v8bf*)(bp0 + 8));
    v16bf B1 = cat8(*(const v8bf*)bp1, *(const v8bf*)(bp1 + 8));
    acc00 = wmma_bf16(A0, B0, acc00);
    acc01 = wmma_bf16(A0, B1, acc01);
    acc10 = wmma_bf16(A1, B0, acc10);
    acc11 = wmma_bf16(A1, B1, acc11);
  }

  const int gn = n0 + nh * 32 + l16;
  const float bv0 = bias[gn], bv1 = bias[gn + 16];
  const int gm0 = m0 + mt * 32 + 8 * hl;
  const int gm1 = gm0 + 16;
#pragma unroll
  for (int r = 0; r < 8; r++) {
    C[(size_t)(gm0 + r) * N + gn]      = acc00[r] + bv0;
    C[(size_t)(gm0 + r) * N + gn + 16] = acc01[r] + bv1;
    C[(size_t)(gm1 + r) * N + gn]      = acc10[r] + bv0;
    C[(size_t)(gm1 + r) * N + gn + 16] = acc11[r] + bv1;
  }
}

// ---------------------------------------------------------------------------
// Gate kernel: one block per (b,t) row. Fuses dwconv5, scalar gates, m_D.
// Writes blended Q/K/V as bf16 and c2 = 2/m_D^2.
// ---------------------------------------------------------------------------
__global__ __launch_bounds__(256) void gate_kernel(
    const float* __restrict__ Qf, const float* __restrict__ Kf, const float* __restrict__ Vf,
    const float* __restrict__ key,
    const float* __restrict__ Wcq, const float* __restrict__ Wck, const float* __restrict__ Wcv,
    const float* __restrict__ Wgq, const float* __restrict__ bgq,
    const float* __restrict__ Wgk, const float* __restrict__ bgk,
    const float* __restrict__ Wgv, const float* __restrict__ bgv,
    const float* __restrict__ WmD, const float* __restrict__ bmD,
    bf16* __restrict__ Qb, bf16* __restrict__ Kb, bf16* __restrict__ Vb,
    float* __restrict__ c2out)
{
  __shared__ float red[256];
  const int row = blockIdx.x;
  const int b = row / TT, t = row % TT;
  const int tid = threadIdx.x;

  float q[2], k[2], v[2], qc[2], kc[2], vc[2];
#pragma unroll
  for (int i = 0; i < 2; i++) {
    const int c = tid + i * 256;
    const size_t off = (size_t)row * SZ + c;
    q[i] = Qf[off]; k[i] = Kf[off]; v[i] = Vf[off];
    float aq = 0.f, ak = 0.f, av = 0.f;
#pragma unroll
    for (int j = 0; j < 5; j++) {
      const int ts = t + j - 2;
      if (ts >= 0 && ts < TT) {
        const float x = key[((size_t)b * TT + ts) * SZ + c];
        aq = fmaf(x, Wcq[c * 5 + j], aq);
        ak = fmaf(x, Wck[c * 5 + j], ak);
        av = fmaf(x, Wcv[c * 5 + j], av);
      }
    }
    qc[i] = aq; kc[i] = ak; vc[i] = av;
  }

  auto blockReduce = [&](float x) -> float {
    red[tid] = x; __syncthreads();
    for (int s = 128; s > 0; s >>= 1) {
      if (tid < s) red[tid] += red[tid + s];
      __syncthreads();
    }
    const float r = red[0];
    __syncthreads();
    return r;
  };

  float p;
  p = 0.f;
#pragma unroll
  for (int i = 0; i < 2; i++) { const int c = tid + i*256; p += q[i]*Wgq[c] + qc[i]*Wgq[SZ + c]; }
  { const float g = 1.f/(1.f + __expf(-(blockReduce(p) + bgq[0])));
#pragma unroll
    for (int i = 0; i < 2; i++) q[i] = (1.f - g)*q[i] + g*qc[i]; }

  p = 0.f;
#pragma unroll
  for (int i = 0; i < 2; i++) { const int c = tid + i*256; p += k[i]*Wgk[c] + kc[i]*Wgk[SZ + c]; }
  { const float g = 1.f/(1.f + __expf(-(blockReduce(p) + bgk[0])));
#pragma unroll
    for (int i = 0; i < 2; i++) k[i] = (1.f - g)*k[i] + g*kc[i]; }

  p = 0.f;
#pragma unroll
  for (int i = 0; i < 2; i++) { const int c = tid + i*256; p += v[i]*Wgv[c] + vc[i]*Wgv[SZ + c]; }
  { const float g = 1.f/(1.f + __expf(-(blockReduce(p) + bgv[0])));
#pragma unroll
    for (int i = 0; i < 2; i++) v[i] = (1.f - g)*v[i] + g*vc[i]; }

  p = 0.f;
#pragma unroll
  for (int i = 0; i < 2; i++) p += q[i] * WmD[tid + i*256];
  const float offv = blockReduce(p) + bmD[0];
  const float mD = 6.3f + 2.8f * tanhf(offv * 0.5f);     // D0 + 2*STD*tanh(off/GAMMA)
  if (tid == 0) c2out[row] = 2.0f / (mD * mD);

#pragma unroll
  for (int i = 0; i < 2; i++) {
    const size_t off = (size_t)row * SZ + tid + i * 256;
    Qb[off] = f2bf(q[i]); Kb[off] = f2bf(k[i]); Vb[off] = f2bf(v[i]);
  }
}

// ---------------------------------------------------------------------------
// One-shot V transpose: Vt2[b][c][t] = Vb[b][t][c]  (per-batch 512x512, bf16).
// Done once so attention PV B-fragments are contiguous global reads.
// ---------------------------------------------------------------------------
__global__ __launch_bounds__(256) void transpose_v_kernel(
    const bf16* __restrict__ Vb, bf16* __restrict__ Vt2)
{
  __shared__ bf16 tile[64][72];
  const int b  = blockIdx.z;
  const int t0 = blockIdx.x * 64;    // time tile
  const int c0 = blockIdx.y * 64;    // channel tile
  const int tid = threadIdx.x;
  for (int i = tid; i < 64 * 64; i += 256) {
    const int r = i >> 6, c = i & 63;
    tile[r][c] = Vb[((size_t)(b * TT + t0 + r)) * SZ + c0 + c];
  }
  __syncthreads();
  for (int i = tid; i < 64 * 64; i += 256) {
    const int c = i >> 6, r = i & 63;
    Vt2[((size_t)b * SZ + c0 + c) * TT + t0 + r] = tile[r][c];
  }
}

// ---------------------------------------------------------------------------
// Attention: one block per (b, h, 16-query tile). Scores kept in LDS only.
// Emits ctx as bf16 so the output projection reads bf16.
// ---------------------------------------------------------------------------
union  __align__(16) AttnRegion0 { float Sbuf[16 * 520]; float pvred[1024]; };
struct __align__(16) AttnSmem {
  AttnRegion0 r0;         // scores (phases 3-4), then cross-wave PV reduction
  bf16  Pbuf[16 * 528];   // normalized attention probs (bf16)
  bf16  Qs[16 * 72];      // scaled Q tile
  float Rbuf[16 * 34];    // qh . rpe_k[r]  for r = 0..32
  float c2s[16];
  float sLo[16];
  float sHi[16];
};

__global__ __launch_bounds__(256) void attn_kernel(
    const bf16* __restrict__ Qb, const bf16* __restrict__ Kb,
    const bf16* __restrict__ Vt2,
    const unsigned char* __restrict__ mask, const float* __restrict__ rpe,
    const float* __restrict__ c2, bf16* __restrict__ ctx)
{
  __shared__ AttnSmem sm;
  const int qt = blockIdx.x, h = blockIdx.y, b = blockIdx.z;
  const int q0 = qt * 16;
  const int tid = threadIdx.x, lane = tid & 31, w = tid >> 5;
  const int hl = lane >> 4, l16 = lane & 15;

  // Phase 1: stage scaled Q tile (qh = q * 0.125), c2 per row
  for (int i = tid; i < 16 * 64; i += 256) {
    const int m = i >> 6, d = i & 63;
    const float x = bf2f(Qb[((size_t)(b * TT + q0 + m)) * SZ + h * HD + d]) * 0.125f;
    sm.Qs[m * 72 + d] = f2bf(x);
  }
  if (tid < 16) sm.c2s[tid] = c2[b * TT + q0 + tid];
  __syncthreads();

  // Phase 2: R[m][r] = qh[m] . rpe_k[r]   (33 clipped offsets)
  for (int i = tid; i < 16 * 33; i += 256) {
    const int m = i / 33, r = i % 33;
    float acc = 0.f;
    for (int d = 0; d < HD; d++)
      acc = fmaf(bf2f(sm.Qs[m * 72 + d]), rpe[r * (2 * HD) + d], acc);
    sm.Rbuf[m * 34 + r] = acc;
  }

  // Phase 3: S = Qs @ Kh^T via WMMA; B-frags straight from global (d contiguous)
  {
    const bf16* ap = sm.Qs + l16 * 72;
    v16bf A0 = cat8(*(const v8bf*)(ap + 16 * hl),      *(const v8bf*)(ap + 16 * hl + 8));
    v16bf A1 = cat8(*(const v8bf*)(ap + 32 + 16 * hl), *(const v8bf*)(ap + 32 + 16 * hl + 8));
    const int n0 = w * 64;
#pragma unroll
    for (int nt = 0; nt < 4; nt++) {
      const int np = n0 + nt * 16 + l16;          // key position for this lane's column
      const bf16* kp = Kb + ((size_t)(b * TT + np)) * SZ + h * HD;
      v16bf B0 = cat8(*(const v8bf*)(kp + 16 * hl),      *(const v8bf*)(kp + 16 * hl + 8));
      v16bf B1 = cat8(*(const v8bf*)(kp + 32 + 16 * hl), *(const v8bf*)(kp + 32 + 16 * hl + 8));
      v8f acc = {};
      acc = wmma_bf16(A0, B0, acc);
      acc = wmma_bf16(A1, B1, acc);
#pragma unroll
      for (int r = 0; r < 8; r++)
        sm.r0.Sbuf[(r + 8 * hl) * 520 + n0 + nt * 16 + l16] = acc[r];
    }
  }
  __syncthreads();

  // Phase 4: bias + mask + softmax (2 rows per wave, wave32 shfl reductions)
#pragma unroll
  for (int rr = 0; rr < 2; rr++) {
    const int m = w * 2 + rr;
    const int qpos = q0 + m;
    const float c2v = sm.c2s[m];
    float s[16];
    float mx = -3.0e38f;
#pragma unroll
    for (int i = 0; i < 16; i++) {
      const int kpos = lane + 32 * i;
      float val = sm.r0.Sbuf[m * 520 + kpos];
      const int dci = kpos - qpos;
      const int rd = dci < -16 ? 0 : (dci > 16 ? 32 : dci + 16);
      val += sm.Rbuf[m * 34 + rd];
      const float dist = (float)(qpos - kpos);
      val -= dist * dist * c2v;
      if (mask[(size_t)b * TT + kpos]) val = -3.0e38f;
      s[i] = val;
      mx = fmaxf(mx, val);
    }
#pragma unroll
    for (int o = 16; o > 0; o >>= 1) mx = fmaxf(mx, __shfl_xor(mx, o, 32));
    float sum = 0.f, lo = 0.f, hi = 0.f;
#pragma unroll
    for (int i = 0; i < 16; i++) {
      const float e = __expf(s[i] - mx);
      s[i] = e;
      sum += e;
      const int dci = lane + 32 * i - qpos;
      if (dci <= -16) lo += e;
      else if (dci >= 16) hi += e;
    }
#pragma unroll
    for (int o = 16; o > 0; o >>= 1) {
      sum += __shfl_xor(sum, o, 32);
      lo  += __shfl_xor(lo,  o, 32);
      hi  += __shfl_xor(hi,  o, 32);
    }
    const float inv = 1.0f / sum;
#pragma unroll
    for (int i = 0; i < 16; i++)
      sm.Pbuf[m * 528 + lane + 32 * i] = f2bf(s[i] * inv);
    if (lane == 0) { sm.sLo[m] = lo * inv; sm.sHi[m] = hi * inv; }
  }
  __syncthreads();

  // Phase 5: ctx = P @ Vh via WMMA; B-frags contiguous from transposed V.
  //          Waves split K in halves; cross-wave reduce; RPE-v epilogue (bf16).
  {
    const int ntile = w & 3, khalf = w >> 2;
    const int d = ntile * 16 + l16;
    const bf16* vrow = Vt2 + ((size_t)b * SZ + h * HD + d) * TT;  // contiguous kpos
    v8f acc = {};
#pragma unroll
    for (int ks = 0; ks < 8; ks++) {
      const int kg = khalf * 256 + ks * 32;
      const bf16* pp = sm.Pbuf + l16 * 528 + kg;
      v16bf Af = cat8(*(const v8bf*)(pp + 16 * hl), *(const v8bf*)(pp + 16 * hl + 8));
      v16bf Bf = cat8(*(const v8bf*)(vrow + kg + 16 * hl),
                      *(const v8bf*)(vrow + kg + 16 * hl + 8));
      acc = wmma_bf16(Af, Bf, acc);
    }
    if (khalf == 1) {
#pragma unroll
      for (int r = 0; r < 8; r++) sm.r0.pvred[ntile * 256 + r * 32 + lane] = acc[r];
    }
    __syncthreads();
    if (khalf == 0) {
#pragma unroll
      for (int r = 0; r < 8; r++) acc[r] += sm.r0.pvred[ntile * 256 + r * 32 + lane];
      const float* rv = rpe + HD;                 // rpe_v columns
#pragma unroll
      for (int r = 0; r < 8; r++) {
        const int m = r + 8 * hl;
        const int qpos = q0 + m;
        float cx = acc[r];
        cx = fmaf(sm.sLo[m], rv[0 * (2 * HD) + d], cx);
        cx = fmaf(sm.sHi[m], rv[32 * (2 * HD) + d], cx);
        for (int c = -15; c <= 15; c++) {
          const int kp2 = qpos + c;
          if (kp2 >= 0 && kp2 < TT)
            cx = fmaf(bf2f(sm.Pbuf[m * 528 + kp2]), rv[(c + 16) * (2 * HD) + d], cx);
        }
        ctx[((size_t)(b * TT + qpos)) * SZ + h * HD + d] = f2bf(cx);
      }
    }
  }
}

// ---------------------------------------------------------------------------
extern "C" void kernel_launch(void* const* d_in, const int* in_sizes, int n_in,
                              void* d_out, int out_size, void* d_ws, size_t ws_size,
                              hipStream_t stream) {
  const float* key   = (const float*)d_in[0];
  const float* value = (const float*)d_in[1];
  const float* query = (const float*)d_in[2];
  const unsigned char* mask = (const unsigned char*)d_in[3];
  const float* Wq  = (const float*)d_in[4];
  const float* bq  = (const float*)d_in[5];
  const float* Wk  = (const float*)d_in[6];
  const float* bk  = (const float*)d_in[7];
  const float* Wv  = (const float*)d_in[8];
  const float* bv  = (const float*)d_in[9];
  const float* Wcq = (const float*)d_in[10];
  const float* Wck = (const float*)d_in[11];
  const float* Wcv = (const float*)d_in[12];
  const float* Wgq = (const float*)d_in[13];
  const float* bgq = (const float*)d_in[14];
  const float* Wgk = (const float*)d_in[15];
  const float* bgk = (const float*)d_in[16];
  const float* Wgv = (const float*)d_in[17];
  const float* bgv = (const float*)d_in[18];
  const float* WmD = (const float*)d_in[19];
  const float* bmD = (const float*)d_in[20];
  const float* rpe = (const float*)d_in[21];
  const float* Wo  = (const float*)d_in[22];
  const float* bo  = (const float*)d_in[23];

  char* ws = (char*)d_ws;
  const size_t NROW = (size_t)BB * TT;          // 8192
  const size_t F32  = NROW * SZ * 4;            // 16 MiB
  const size_t B16  = NROW * SZ * 2;            // 8 MiB
  const size_t WB16 = (size_t)SZ * SZ * 2;      // 512 KiB

  // Region 0: f32 Q/K/V (dead after gate) — ctx_b aliases its start.
  float* Qf  = (float*)(ws);
  float* Kf  = (float*)(ws + F32);
  float* Vf  = (float*)(ws + 2 * F32);
  bf16*  ctxb = (bf16*)(ws);                    // alias: used only after gate
  // Region 1: bf16 activations (inputs pre-cvt; later overwritten by gated QKV)
  bf16*  Xq  = (bf16*)(ws + 3 * F32);
  bf16*  Xk  = (bf16*)(ws + 3 * F32 + B16);
  bf16*  Xv  = (bf16*)(ws + 3 * F32 + 2 * B16);
  bf16*  Qb  = Xq;                              // alias: Xq dead after proj GEMMs
  bf16*  Kb  = Xk;
  bf16*  Vb  = Xv;
  // Region 2: bf16 weights + transposed V + c2
  bf16*  Wqb = (bf16*)(ws + 3 * F32 + 3 * B16);
  bf16*  Wkb = Wqb + SZ * SZ;
  bf16*  Wvb = Wkb + SZ * SZ;
  bf16*  Wob = Wvb + SZ * SZ;
  bf16*  Vt2 = (bf16*)(ws + 3 * F32 + 3 * B16 + 4 * WB16);
  float* c2  = (float*)(ws + 3 * F32 + 3 * B16 + 4 * WB16 + B16);

  const int nAct = (int)(NROW * SZ);            // 4194304
  const int nWt  = SZ * SZ;                     // 262144
  cvt_bf16_kernel<<<nAct / 2048, 256, 0, stream>>>(query, Xq, nAct);
  cvt_bf16_kernel<<<nAct / 2048, 256, 0, stream>>>(key,   Xk, nAct);
  cvt_bf16_kernel<<<nAct / 2048, 256, 0, stream>>>(value, Xv, nAct);
  cvt_bf16_kernel<<<nWt  / 2048, 256, 0, stream>>>(Wq, Wqb, nWt);
  cvt_bf16_kernel<<<nWt  / 2048, 256, 0, stream>>>(Wk, Wkb, nWt);
  cvt_bf16_kernel<<<nWt  / 2048, 256, 0, stream>>>(Wv, Wvb, nWt);
  cvt_bf16_kernel<<<nWt  / 2048, 256, 0, stream>>>(Wo, Wob, nWt);

  dim3 gg(SZ / 64, (unsigned)(NROW / 128));
  gemm_bf16_kernel<<<gg, 256, 0, stream>>>(Xq, Wqb, bq, Qf, (int)NROW, SZ, SZ);
  gemm_bf16_kernel<<<gg, 256, 0, stream>>>(Xk, Wkb, bk, Kf, (int)NROW, SZ, SZ);
  gemm_bf16_kernel<<<gg, 256, 0, stream>>>(Xv, Wvb, bv, Vf, (int)NROW, SZ, SZ);
  gate_kernel<<<(unsigned)NROW, 256, 0, stream>>>(Qf, Kf, Vf, key,
      Wcq, Wck, Wcv, Wgq, bgq, Wgk, bgk, Wgv, bgv, WmD, bmD, Qb, Kb, Vb, c2);
  transpose_v_kernel<<<dim3(TT / 64, SZ / 64, BB), 256, 0, stream>>>(Vb, Vt2);
  attn_kernel<<<dim3(TT / 16, HH, BB), 256, 0, stream>>>(Qb, Kb, Vt2, mask, rpe, c2, ctxb);
  gemm_bf16_kernel<<<gg, 256, 0, stream>>>(ctxb, Wob, bo, (float*)d_out, (int)NROW, SZ, SZ);
}
